// DeepLSTM_22454089023801
// MI455X (gfx1250) — compile-verified, matrix operand order checked by reading, not run on hardware
//
#include <hip/hip_runtime.h>
#include <stdint.h>

// ---- problem constants (from reference) ----
#define VOCAB 32000
#define SIZE  1024
#define DEPTH 3
#define BATCH 64
#define STEPS 256

#define KTOT   (3 * SIZE)          // 3072 : concat(x, skip, h)
#define NTOT   (4 * SIZE)          // 4096 : gates i,j,f,o
#define KT     (KTOT / 32)         // 96 K-tiles of 32
#define NT     (NTOT / 16)         // 256 N-tiles of 16

typedef __bf16 v16bf __attribute__((ext_vector_type(16)));
typedef float  v8f   __attribute__((ext_vector_type(8)));

union Op32 { uint4 q[2]; v16bf v; };   // 32B WMMA bf16 operand (8 VGPRs)

__device__ __forceinline__ unsigned short f2bf(float f) {
    union { float f; unsigned int u; } x; x.f = f;
    unsigned int r = x.u + 0x7FFFu + ((x.u >> 16) & 1u);   // round-to-nearest-even
    return (unsigned short)(r >> 16);
}

// gfx1250 async copy: global -> LDS, 16B per lane, tracked by ASYNCcnt
__device__ __forceinline__ void async_b128(unsigned int ldsAddr, const void* gaddr) {
    asm volatile("global_load_async_to_lds_b128 %0, %1, off"
                 :: "v"(ldsAddr), "v"((unsigned long long)(uintptr_t)gaddr)
                 : "memory");
}
__device__ __forceinline__ void wait_async0() {
    asm volatile("s_wait_asynccnt 0x0" ::: "memory");
}

// ---------------------------------------------------------------------------
// Weights f32 [DEPTH][3S][4S] row-major  ->  bf16, swizzled into WMMA-B tile
// order bw[layer][nt][kt][lane][16]: each lane's B operand (K=32 x N=16 tile)
// is one contiguous 32-byte load. 75.5 MB total -> L2-resident.
// ---------------------------------------------------------------------------
__global__ void convert_swizzle_weights(const float* __restrict__ Ws,
                                        unsigned short* __restrict__ bw) {
    long tid = (long)blockIdx.x * blockDim.x + threadIdx.x;
    const long total = (long)DEPTH * NT * KT * 32;
    if (tid >= total) return;
    int  lane = (int)(tid & 31);
    long rest = tid >> 5;
    int  kt = (int)(rest % KT);  rest /= KT;
    int  nt = (int)(rest % NT);  rest /= NT;
    int  layer = (int)rest;

    int n     = nt * 16 + (lane & 15);
    int kbase = kt * 32 + (lane >> 4) * 16;
    const float* w = Ws + (long)layer * KTOT * NTOT;

    unsigned short tmp[16];
#pragma unroll
    for (int e = 0; e < 16; ++e)                 // element e -> K = kbase + e
        tmp[e] = f2bf(w[(long)(kbase + e) * NTOT + n]);

    uint4* dst = (uint4*)(bw + ((((long)layer * NT + nt) * KT + kt) * 32 + lane) * 16);
    dst[0] = *(uint4*)&tmp[0];
    dst[1] = *(uint4*)&tmp[8];
}

// Embedding gather to bf16, Xbf[t][b][s]. Off the critical path.
__global__ void gather_x_bf16(const float* __restrict__ emb,
                              const int*   __restrict__ tokens,
                              unsigned short* __restrict__ Xbf) {
    long i = (long)blockIdx.x * blockDim.x + threadIdx.x;   // one thread per 8 elems
    const long total = (long)STEPS * BATCH * (SIZE / 8);
    if (i >= total) return;
    int s8 = (int)(i % (SIZE / 8)) * 8;
    long rest = i / (SIZE / 8);
    int b = (int)(rest % BATCH);
    int t = (int)(rest / BATCH);
    int tok = tokens[b * STEPS + t];
    const float4* src = (const float4*)(emb + (long)tok * SIZE + s8);
    float4 a = src[0], c = src[1];
    unsigned short pk[8] = { f2bf(a.x), f2bf(a.y), f2bf(a.z), f2bf(a.w),
                             f2bf(c.x), f2bf(c.y), f2bf(c.z), f2bf(c.w) };
    *(uint4*)(Xbf + ((long)t * BATCH + b) * SIZE + s8) = *(uint4*)pk;
}

__global__ void zero_words(unsigned int* __restrict__ p, int n) {
    int i = blockIdx.x * blockDim.x + threadIdx.x;
    if (i < n) p[i] = 0u;
}

// ---------------------------------------------------------------------------
// One fused recurrence step for (t, layer).
// grid = 16 blocks (64-wide S slices), block = 128 threads (4 waves).
// Wave w computes gate w's 64x64 pre-activation tile (4x4 WMMA tiles).
// A = [x_t | h[layer-1]@t | h[layer]@t-1], all bf16 in global, staged into a
// double-buffered LDS chunk via global_load_async_to_lds_b128 (ASYNCcnt).
// Gates meet in LDS, then the f32 elementwise cell update.
// ---------------------------------------------------------------------------
__global__ __launch_bounds__(128)
void lstm_step(const unsigned short* __restrict__ bw,
               const float* __restrict__ bs,
               const unsigned short* __restrict__ Xbf,
               const unsigned short* __restrict__ hbprev, // bf16 h at t-1
               unsigned short* __restrict__ hbcur,        // bf16 h at t (r l-1, w l)
               float* __restrict__ hf32,                  // f32 h (output only)
               float* __restrict__ cstate,                // f32 c
               int t, int layer) {
    __shared__ float gbuf[4][64][64];                // 64 KB gate staging
    unsigned short* As = (unsigned short*)gbuf;      // aliased double A buffer:
                                                     // buf b at elems b*2560, 64 rows x 40

    const int tid   = threadIdx.x;
    const int lane  = tid & 31;
    const int wgate = tid >> 5;                      // wave -> gate (i,j,f,o)
    const int s0    = blockIdx.x * 64;               // S slice

    // staging role: thread covers batch row r, 16-wide K half kh
    const int r  = tid >> 1;
    const int kh = (tid & 1) * 16;

    // per-region bf16 source rows (each advances 32 elems per chunk)
    const unsigned short* xrow   = Xbf + ((long)t * BATCH + r) * SIZE + kh;
    const unsigned short* skiprow = (layer > 0)
        ? hbcur + ((long)(layer - 1) * BATCH + r) * SIZE + kh : nullptr;
    const unsigned short* recrow  = hbprev + ((long)layer * BATCH + r) * SIZE + kh;

    const unsigned int ldsBase =
        (unsigned int)(uintptr_t)As + (unsigned int)(r * 80 + kh * 2); // bytes
    // stage chunk kc into LDS buffer (kc&1)
    auto stage = [&](int kc) {
        const int c = kc & 31;            // chunk within region
        const int region = kc >> 5;       // 0=x, 1=skip, 2=recurrent
        unsigned int dst = ldsBase + (unsigned int)((kc & 1) * 5120);
        if (region == 0) {
            const unsigned short* g = xrow + c * 32;
            async_b128(dst, g); async_b128(dst + 16, g + 8);
        } else if (region == 1) {
            if (layer > 0) {
                const unsigned short* g = skiprow + c * 32;
                async_b128(dst, g); async_b128(dst + 16, g + 8);
            } else {
                uint4 z; z.x = z.y = z.z = z.w = 0u;   // layer-0 skip slot = zeros
                *(uint4*)&As[(kc & 1) * 2560 + r * 40 + kh] = z;
            }
        } else {
            const unsigned short* g = recrow + c * 32;
            async_b128(dst, g); async_b128(dst + 16, g + 8);
        }
    };

    v8f acc[4][4];
#pragma unroll
    for (int mt = 0; mt < 4; ++mt)
#pragma unroll
        for (int nti = 0; nti < 4; ++nti)
#pragma unroll
            for (int v = 0; v < 8; ++v) acc[mt][nti][v] = 0.0f;

    stage(0);
    wait_async0();
    __syncthreads();

    for (int kc = 0; kc < KT; ++kc) {
        if (kc + 1 < KT) stage(kc + 1);            // prefetch next chunk (other buffer)

        const int abase = (kc & 1) * 2560;
        Op32 a[4], b[4];
#pragma unroll
        for (int mt = 0; mt < 4; ++mt) {
            // A 16x32 bf16: lane<16 holds row(lane) K 0..15; lane>=16 K 16..31
            const uint4* p = (const uint4*)
                &As[abase + (mt * 16 + (lane & 15)) * 40 + (lane >> 4) * 16];
            a[mt].q[0] = p[0]; a[mt].q[1] = p[1];
        }
#pragma unroll
        for (int nti = 0; nti < 4; ++nti) {
            int nt = wgate * 64 + blockIdx.x * 4 + nti;  // col = gate*1024 + s0 + nti*16
            const uint4* p = (const uint4*)(bw +
                ((((long)layer * NT + nt) * KT + kc) * 32 + lane) * 16);
            b[nti].q[0] = p[0]; b[nti].q[1] = p[1];
        }
#pragma unroll
        for (int mt = 0; mt < 4; ++mt)
#pragma unroll
            for (int nti = 0; nti < 4; ++nti)
                acc[mt][nti] = __builtin_amdgcn_wmma_f32_16x16x32_bf16(
                    false, a[mt].v, false, b[nti].v,
                    (short)0, acc[mt][nti], false, false);

        wait_async0();      // next chunk resident in LDS
        __syncthreads();    // all waves done reading current chunk / see new one
    }

    // ---- spill gate pre-activations to LDS ----
#pragma unroll
    for (int mt = 0; mt < 4; ++mt)
#pragma unroll
        for (int nti = 0; nti < 4; ++nti)
#pragma unroll
            for (int v = 0; v < 8; ++v)
                gbuf[wgate][mt * 16 + (lane >> 4) * 8 + v][nti * 16 + (lane & 15)]
                    = acc[mt][nti][v];
    __syncthreads();

    // ---- elementwise LSTM cell update (f32) ----
    const float* bias = bs + (long)layer * NTOT;
    for (int p = tid; p < 64 * 64; p += 128) {
        int s = p & 63, b = p >> 6;
        int scol = s0 + s;
        float iv = gbuf[0][b][s] + bias[0 * SIZE + scol];
        float jv = gbuf[1][b][s] + bias[1 * SIZE + scol];
        float fv = gbuf[2][b][s] + bias[2 * SIZE + scol];
        float ov = gbuf[3][b][s] + bias[3 * SIZE + scol];
        long sidx = ((long)layer * BATCH + b) * SIZE + scol;
        float c_old = cstate[sidx];
        float si = 1.0f / (1.0f + __expf(-iv));
        float sf = 1.0f / (1.0f + __expf(-fv));
        float so = 1.0f / (1.0f + __expf(-ov));
        float c_new = c_old * sf + si * tanhf(jv);
        float h_new = tanhf(c_new) * so;
        cstate[sidx] = c_new;
        hbcur[sidx]  = f2bf(h_new);   // bf16 for the next matmuls
        hf32[sidx]   = h_new;         // f32 for the final output
    }
}

// out[b][layer*S + s] = h_final[layer][b][s]
__global__ void finalize_out(const float* __restrict__ h, float* __restrict__ out) {
    int i = blockIdx.x * blockDim.x + threadIdx.x;
    if (i >= DEPTH * BATCH * SIZE) return;
    int s = i % SIZE; int rest = i / SIZE;
    int b = rest % BATCH; int layer = rest / BATCH;
    out[(long)b * (DEPTH * SIZE) + layer * SIZE + s] = h[i];
}

extern "C" void kernel_launch(void* const* d_in, const int* in_sizes, int n_in,
                              void* d_out, int out_size, void* d_ws, size_t ws_size,
                              hipStream_t stream) {
    const int*   tokens = (const int*)  d_in[0];   // [BATCH, STEPS] int32
    const float* emb    = (const float*)d_in[1];   // [VOCAB, SIZE]
    const float* Ws     = (const float*)d_in[2];   // [DEPTH, 3S, 4S]
    const float* bs     = (const float*)d_in[3];   // [DEPTH, 4S]

    // scratch layout: bf16 weights | bf16 X | bf16 H0 | bf16 H1 | f32 C | f32 H
    char* wsp = (char*)d_ws;
    unsigned short* bw  = (unsigned short*)wsp;                 // 75,497,472 B
    wsp += (size_t)DEPTH * KTOT * NTOT * 2;
    unsigned short* Xbf = (unsigned short*)wsp;                 // 33,554,432 B
    wsp += (size_t)STEPS * BATCH * SIZE * 2;
    unsigned short* Hb0 = (unsigned short*)wsp;                 // 393,216 B
    wsp += (size_t)DEPTH * BATCH * SIZE * 2;
    unsigned short* Hb1 = (unsigned short*)wsp;                 // 393,216 B
    wsp += (size_t)DEPTH * BATCH * SIZE * 2;
    float* C    = (float*)wsp;                                  // 786,432 B
    wsp += (size_t)DEPTH * BATCH * SIZE * 4;
    float* Hf32 = (float*)wsp;                                  // 786,432 B

    // zero Hb0 | Hb1 | C (contiguous words)
    int zN = (int)(((size_t)DEPTH * BATCH * SIZE * 2 * 2 +
                    (size_t)DEPTH * BATCH * SIZE * 4) / 4);
    zero_words<<<(zN + 255) / 256, 256, 0, stream>>>((unsigned int*)Hb0, zN);

    // one-time bf16 weight swizzle + embedding gather (parallel, off chain)
    long cw = (long)DEPTH * NT * KT * 32;
    convert_swizzle_weights<<<(int)((cw + 255) / 256), 256, 0, stream>>>(Ws, bw);
    long gx = (long)STEPS * BATCH * (SIZE / 8);
    gather_x_bf16<<<(int)((gx + 255) / 256), 256, 0, stream>>>(emb, tokens, Xbf);

    // sequential recurrence: ping-pong bf16 h buffers per timestep
    for (int t = 0; t < STEPS; ++t) {
        unsigned short* hbprev = (t & 1) ? Hb1 : Hb0;
        unsigned short* hbcur  = (t & 1) ? Hb0 : Hb1;
        for (int layer = 0; layer < DEPTH; ++layer)
            lstm_step<<<16, 128, 0, stream>>>(bw, bs, Xbf, hbprev, hbcur,
                                              Hf32, C, t, layer);
    }
    int fN = DEPTH * BATCH * SIZE;
    finalize_out<<<(fN + 255) / 256, 256, 0, stream>>>(Hf32, (float*)d_out);
}
